// AdaptiveMemoryBank_13932873908445
// MI455X (gfx1250) — matrix operand order repaired and checked by reading, not compiled, not run on hardware
//
#include <hip/hip_runtime.h>

typedef __attribute__((ext_vector_type(16))) __bf16 v16bf;
typedef __attribute__((ext_vector_type(8)))  float  v8f;

union Frag {
  v16bf v;
  uint4 u4[2];
};

// Split f32 into bf16 hi (truncated) + bf16 lo (exact residual, truncated).
// a ~= hi + lo with |err| ~ 2^-16 * |a| after dropping lo*lo cross term.
__device__ __forceinline__ void bsplit(float x, unsigned int& h, unsigned int& l) {
  unsigned int u  = __float_as_uint(x);
  unsigned int hu = u & 0xFFFF0000u;
  float lf = x - __uint_as_float(hu);   // exactly representable
  h = hu >> 16;
  l = __float_as_uint(lf) >> 16;
}

#define TM 64
#define TN 256
#define TK 32
#define LDT 40   // ushort stride of LDS tiles (32 + 8 pad): conflict-free b128 reads

// C[M,N] = epi(A[M,K] x W[N,K]^T + bias), f32 in/out, bf16x3 WMMA core.
// SCALE: multiply A element at col k by (k < K/2 ? scale_lo : scale_hi)[row]
// EPI: 0 = store, 1 = relu+store, 2 = store only rows with selected[row]==branch
template<bool SCALE, int EPI>
__global__ __launch_bounds__(256)
void gemm_bf16x3(const float* __restrict__ A, const float* __restrict__ W,
                 const float* __restrict__ bias, float* __restrict__ C,
                 int M, int N, int K,
                 const float* __restrict__ scale_lo,
                 const float* __restrict__ scale_hi,
                 const int* __restrict__ selected, int branch)
{
  __shared__ __align__(16) unsigned short AsH[TM * LDT];
  __shared__ __align__(16) unsigned short AsL[TM * LDT];
  __shared__ __align__(16) unsigned short BsH[TN * LDT];
  __shared__ __align__(16) unsigned short BsL[TN * LDT];

  const int t      = threadIdx.x;
  const int lane16 = t & 15;          // == (lane & 15)
  const int laneHi = (t >> 4) & 1;    // == (lane >> 4)
  const int wid    = t >> 5;          // wave id 0..7
  const int mw     = wid >> 2;        // 0..1 : wave M offset = mw*32
  const int nw     = wid & 3;         // 0..3 : wave N offset = nw*64
  const int mBase  = blockIdx.y * TM;
  const int nBase  = blockIdx.x * TN;

  v8f acc[2][4];
  #pragma unroll
  for (int i = 0; i < 2; ++i)
    #pragma unroll
    for (int j = 0; j < 4; ++j)
      #pragma unroll
      for (int e = 0; e < 8; ++e) acc[i][j][e] = 0.0f;

  for (int k0 = 0; k0 < K; k0 += TK) {
    // ---- stage A tile (64 x 32 f32 -> split bf16 hi/lo) : 512 float4 ----
    #pragma unroll
    for (int it = 0; it < 2; ++it) {
      int idx4 = t + it * 256;
      int r = idx4 >> 3;
      int c = (idx4 & 7) << 2;
      float4 vv = *(const float4*)(A + (size_t)(mBase + r) * K + (k0 + c));
      if (SCALE) {
        float s = (k0 < (K >> 1)) ? scale_lo[mBase + r] : scale_hi[mBase + r];
        vv.x *= s; vv.y *= s; vv.z *= s; vv.w *= s;
      }
      unsigned int h0,h1,h2,h3,l0,l1,l2,l3;
      bsplit(vv.x,h0,l0); bsplit(vv.y,h1,l1); bsplit(vv.z,h2,l2); bsplit(vv.w,h3,l3);
      uint2 hh, ll;
      hh.x = h0 | (h1 << 16); hh.y = h2 | (h3 << 16);
      ll.x = l0 | (l1 << 16); ll.y = l2 | (l3 << 16);
      *(uint2*)(AsH + r * LDT + c) = hh;
      *(uint2*)(AsL + r * LDT + c) = ll;
    }
    // ---- stage B tile (256 x 32 f32 from W rows) : 2048 float4 ----
    #pragma unroll
    for (int it = 0; it < 8; ++it) {
      int idx4 = t + it * 256;
      int r = idx4 >> 3;
      int c = (idx4 & 7) << 2;
      float4 vv = *(const float4*)(W + (size_t)(nBase + r) * K + (k0 + c));
      unsigned int h0,h1,h2,h3,l0,l1,l2,l3;
      bsplit(vv.x,h0,l0); bsplit(vv.y,h1,l1); bsplit(vv.z,h2,l2); bsplit(vv.w,h3,l3);
      uint2 hh, ll;
      hh.x = h0 | (h1 << 16); hh.y = h2 | (h3 << 16);
      ll.x = l0 | (l1 << 16); ll.y = l2 | (l3 << 16);
      *(uint2*)(BsH + r * LDT + c) = hh;
      *(uint2*)(BsL + r * LDT + c) = ll;
    }
    __syncthreads();

    // ---- build fragments per ISA 7.12.2 layouts and issue WMMAs ----
    // A (16x32 bf16): lane row = lane16; halves 0..7 = K[8*laneHi .. +7],
    //                 halves 8..15 = K[16 + 8*laneHi .. +7]
    Frag ah[2], al[2];
    #pragma unroll
    for (int sm = 0; sm < 2; ++sm) {
      int off = (mw * 32 + sm * 16 + lane16) * LDT + laneHi * 8;
      ah[sm].u4[0] = *(const uint4*)(AsH + off);
      ah[sm].u4[1] = *(const uint4*)(AsH + off + 16);
      al[sm].u4[0] = *(const uint4*)(AsL + off);
      al[sm].u4[1] = *(const uint4*)(AsL + off + 16);
    }
    // B (32x16 bf16): lane col = lane16 (a W row); halves 0..7 = K[8*laneHi .. +7],
    //                 halves 8..15 = K[16 + 8*laneHi .. +7]
    #pragma unroll
    for (int sn = 0; sn < 4; ++sn) {
      int off = (nw * 64 + sn * 16 + lane16) * LDT + laneHi * 8;
      Frag bh, bl;
      bh.u4[0] = *(const uint4*)(BsH + off);
      bh.u4[1] = *(const uint4*)(BsH + off + 16);
      bl.u4[0] = *(const uint4*)(BsL + off);
      bl.u4[1] = *(const uint4*)(BsL + off + 16);
      #pragma unroll
      for (int sm = 0; sm < 2; ++sm) {
        acc[sm][sn] = __builtin_amdgcn_wmma_f32_16x16x32_bf16(
            false, ah[sm].v, false, bh.v, (short)0, acc[sm][sn], false, false);
        acc[sm][sn] = __builtin_amdgcn_wmma_f32_16x16x32_bf16(
            false, ah[sm].v, false, bl.v, (short)0, acc[sm][sn], false, false);
        acc[sm][sn] = __builtin_amdgcn_wmma_f32_16x16x32_bf16(
            false, al[sm].v, false, bh.v, (short)0, acc[sm][sn], false, false);
      }
    }
    __syncthreads();
  }

  // ---- epilogue: D layout = lane col (lane16), rows = 8*laneHi + vgpr ----
  #pragma unroll
  for (int sn = 0; sn < 4; ++sn) {
    int col = nBase + nw * 64 + sn * 16 + lane16;
    float bv = bias[col];
    #pragma unroll
    for (int sm = 0; sm < 2; ++sm) {
      int rbase = mBase + mw * 32 + sm * 16 + laneHi * 8;
      #pragma unroll
      for (int e = 0; e < 8; ++e) {
        int row = rbase + e;
        float v = acc[sm][sn][e] + bv;
        if (EPI == 1) v = v > 0.0f ? v : 0.0f;
        if (EPI == 2) { if (selected[row] != branch) continue; }
        C[(size_t)row * N + col] = v;
      }
    }
  }
}

// logits = hid @ sel2_W^T + b ; selected = argmax  (3-wide, VALU wave-reduce)
__global__ __launch_bounds__(256)
void selector_kernel(const float* __restrict__ hid, const float* __restrict__ W,
                     const float* __restrict__ b, int* __restrict__ selected,
                     int Ntok, int Kd)
{
  int wave = (int)((blockIdx.x * (size_t)blockDim.x + threadIdx.x) >> 5);
  int lane = threadIdx.x & 31;
  if (wave >= Ntok) return;
  const float* hrow = hid + (size_t)wave * Kd;
  float a0 = 0.f, a1 = 0.f, a2 = 0.f;
  for (int i = lane; i < Kd; i += 32) {
    float hv = hrow[i];
    a0 += hv * W[0 * Kd + i];
    a1 += hv * W[1 * Kd + i];
    a2 += hv * W[2 * Kd + i];
  }
  #pragma unroll
  for (int off = 16; off > 0; off >>= 1) {
    a0 += __shfl_xor(a0, off, 32);
    a1 += __shfl_xor(a1, off, 32);
    a2 += __shfl_xor(a2, off, 32);
  }
  if (lane == 0) {
    float l0 = a0 + b[0], l1 = a1 + b[1], l2 = a2 + b[2];
    int s = 0; float m = l0;                 // argmax(softmax(x)) == argmax(x)
    if (l1 > m) { m = l1; s = 1; }
    if (l2 > m) { m = l2; s = 2; }
    selected[wave] = s;
  }
}

// rows with selected==0 pass hidden_states through
__global__ __launch_bounds__(256)
void copy_sel0(const float* __restrict__ X, const int* __restrict__ sel,
               float* __restrict__ out, size_t total4)
{
  size_t i = blockIdx.x * (size_t)blockDim.x + threadIdx.x;
  if (i >= total4) return;
  int row = (int)(i >> 9);                   // 512 float4 per 2048-wide row
  if (sel[row] == 0) {
    ((float4*)out)[i] = ((const float4*)X)[i];
  }
}

extern "C" void kernel_launch(void* const* d_in, const int* in_sizes, int n_in,
                              void* d_out, int out_size, void* d_ws, size_t ws_size,
                              hipStream_t stream)
{
  const float* X    = (const float*)d_in[0];
  const float* freq = (const float*)d_in[1];
  const float* imp  = (const float*)d_in[2];
  const float* c1W  = (const float*)d_in[3];  const float* c1b = (const float*)d_in[4];
  const float* a1W  = (const float*)d_in[5];  const float* a1b = (const float*)d_in[6];
  const float* d1W  = (const float*)d_in[7];  const float* d1b = (const float*)d_in[8];
  const float* c2W  = (const float*)d_in[9];  const float* c2b = (const float*)d_in[10];
  const float* a2W  = (const float*)d_in[11]; const float* a2b = (const float*)d_in[12];
  const float* d2W  = (const float*)d_in[13]; const float* d2b = (const float*)d_in[14];
  const float* s1W  = (const float*)d_in[15]; const float* s1b = (const float*)d_in[16];
  const float* s2W  = (const float*)d_in[17]; const float* s2b = (const float*)d_in[18];
  float* out = (float*)d_out;

  const int M = 4 * 4096;   // tokens
  const int H = 2048;

  char* ws    = (char*)d_ws;
  float* buf0 = (float*)ws;                               // <= M*1024 f32 (64MB)
  float* buf1 = (float*)(ws + (size_t)M * 1024 * 4);      // <= M*1024 f32 (64MB)
  int*   sel  = (int*)  (ws + (size_t)M * 1024 * 4 * 2);  // M ints

  dim3 blk(256);
  auto grid = [&](int n) { return dim3(n / TN, M / TM); };

  // 1) hid = relu((scaled X) @ sel1_W^T + b)      [M,512]
  gemm_bf16x3<true, 1><<<grid(512), blk, 0, stream>>>(
      X, s1W, s1b, buf0, M, 512, H, freq, imp, nullptr, 0);
  // 2) selected = argmax(hid @ sel2_W^T + b)
  selector_kernel<<<dim3(M / 8), blk, 0, stream>>>(buf0, s2W, s2b, sel, M, 512);
  // 3) branch1: compress -> adapt -> decompress (stores only sel==1 rows)
  gemm_bf16x3<false, 0><<<grid(1024), blk, 0, stream>>>(
      X, c1W, c1b, buf1, M, 1024, H, nullptr, nullptr, nullptr, 0);
  gemm_bf16x3<false, 0><<<grid(1024), blk, 0, stream>>>(
      buf1, a1W, a1b, buf0, M, 1024, 1024, nullptr, nullptr, nullptr, 0);
  gemm_bf16x3<false, 2><<<grid(2048), blk, 0, stream>>>(
      buf0, d1W, d1b, out, M, 2048, 1024, nullptr, nullptr, sel, 1);
  // 4) branch2 (stores only sel==2 rows)
  gemm_bf16x3<false, 0><<<grid(512), blk, 0, stream>>>(
      X, c2W, c2b, buf1, M, 512, H, nullptr, nullptr, nullptr, 0);
  gemm_bf16x3<false, 0><<<grid(512), blk, 0, stream>>>(
      buf1, a2W, a2b, buf0, M, 512, 512, nullptr, nullptr, nullptr, 0);
  gemm_bf16x3<false, 2><<<grid(2048), blk, 0, stream>>>(
      buf0, d2W, d2b, out, M, 2048, 512, nullptr, nullptr, sel, 2);
  // 5) passthrough rows (sel==0)
  size_t total4 = (size_t)M * H / 4;
  copy_sel0<<<dim3((unsigned)((total4 + 255) / 256)), blk, 0, stream>>>(
      X, sel, out, total4);
}
